// sparseSpatialAttention_6038724018671
// MI455X (gfx1250) — matrix operand order repaired and checked by reading, not compiled, not run on hardware
//
#include <hip/hip_runtime.h>
#include <stdint.h>

// Problem constants (from reference): B=8, T=12, N=512, D=64, NN=16, M_SAMPLES=45
#define BT   96
#define Nn   512
#define Dd   64
#define NNb  16
#define MS   45
#define MSP  48     // padded row count for 16-row WMMA tiles
#define LDSTR 68    // padded LDS row stride (floats) for 16x64 tiles

typedef __attribute__((ext_vector_type(16))) _Float16 v16h;
typedef __attribute__((ext_vector_type(8)))  float    v8f;
typedef __attribute__((ext_vector_type(4)))  int      v4i;

typedef __attribute__((address_space(1))) v4i* gptr_v4i;
typedef __attribute__((address_space(3))) v4i* lptr_v4i;

static __device__ __forceinline__ v8f wmma16(v16h a, v16h b, v8f c) {
  // D = A(16x32 f16) x B(32x16 f16) + C(16x16 f32)
  return __builtin_amdgcn_wmma_f32_16x16x32_f16(false, a, false, b, (short)0, c,
                                                false, false);
}

// B fragment (32x16): B[k][n] = W[(k0+k)*ldb + n0+n]. Per ISA layout a lane
// holds one column: col = lane&15 (+n0), K base = k0 + (lane>=16 ? 8 : 0),
// elements e<8 -> K = Kbase+e, elements 8+e -> K = Kbase+16+e.
static __device__ __forceinline__ v16h load_b_rm(const float* W, int ldb, int k0, int n0) {
  const int lane = threadIdx.x & 31;
  const int col  = n0 + (lane & 15);
  const int kb   = k0 + ((lane >> 4) << 3);
  v16h b;
#pragma unroll
  for (int e = 0; e < 8; ++e) {
    b[e]     = (_Float16)W[(size_t)(kb + e) * ldb + col];
    b[8 + e] = (_Float16)W[(size_t)(kb + 16 + e) * ldb + col];
  }
  return b;
}

// ---------------------------------------------------------------------------
// Async staging of a 64KB f16 slab (512x64) from global into LDS.
// Uses GLOBAL_LOAD_ASYNC_TO_LDS_B128 (ASYNCcnt) when the toolchain exposes it.
static __device__ __forceinline__ void async_stage_slab(const _Float16* __restrict__ gsrc,
                                                        _Float16* lds, int tid, int nthreads) {
#if __has_builtin(__builtin_amdgcn_global_load_async_to_lds_b128)
  gptr_v4i gp = (gptr_v4i)gsrc;
  lptr_v4i lp = (lptr_v4i)lds;
  for (int i = tid; i < (Nn * Dd * 2) / 16; i += nthreads)
    __builtin_amdgcn_global_load_async_to_lds_b128(gp + i, lp + i, 0, 0);
#else
  const float* gf = (const float*)gsrc;   // copy as dwords
  float* lf = (float*)lds;
  for (int i = tid; i < (Nn * Dd) / 2; i += nthreads) lf[i] = gf[i];
#endif
}

static __device__ __forceinline__ void wait_async_done() {
#if __has_builtin(__builtin_amdgcn_s_wait_asynccnt)
  __builtin_amdgcn_s_wait_asynccnt(0);
#else
  asm volatile("s_wait_asynccnt 0x0" ::: "memory");
#endif
}

// ---------------------------------------------------------------------------
// Kernel 1: Q/K/V = (x + ev) @ W{q,k,v} + b   — one wave per 16-row tile.
// Also emits f16 shadow copies (Qh/Kh/Vh) for the WMMA attention stage.
// grid (N/16, BT), block 32.
__global__ __launch_bounds__(32)
void qkv_wmma(const float* __restrict__ x, const float* __restrict__ eigvec,
              const float* __restrict__ eigval,
              const float* __restrict__ Wq, const float* __restrict__ bq,
              const float* __restrict__ Wk, const float* __restrict__ bk,
              const float* __restrict__ Wv, const float* __restrict__ bv,
              float* __restrict__ Qb, float* __restrict__ Kb, float* __restrict__ Vb,
              _Float16* __restrict__ Qh, _Float16* __restrict__ Kh, _Float16* __restrict__ Vh)
{
  const int bt = blockIdx.y;
  const int n0 = blockIdx.x * 16;
  const int lane  = threadIdx.x;
  const int rloc  = lane & 15;
  const int khalf = lane >> 4;

  // A fragments of x_ = x + eigvec[d,n]*eigval[d], rows n0..n0+15, K=0..63.
  const int n = n0 + rloc;
  const float* xr = x + ((size_t)bt * Nn + n) * Dd;
  const int kb = khalf * 8;
  v16h a0, a1;
#pragma unroll
  for (int e = 0; e < 8; ++e) {
    int k;
    k = kb + e;           a0[e]     = (_Float16)(xr[k] + eigvec[(size_t)k * Nn + n] * eigval[k]);
    k = kb + 16 + e;      a0[8 + e] = (_Float16)(xr[k] + eigvec[(size_t)k * Nn + n] * eigval[k]);
    k = 32 + kb + e;      a1[e]     = (_Float16)(xr[k] + eigvec[(size_t)k * Nn + n] * eigval[k]);
    k = 32 + kb + 16 + e; a1[8 + e] = (_Float16)(xr[k] + eigvec[(size_t)k * Nn + n] * eigval[k]);
  }

  const float*    Ws[3] = {Wq, Wk, Wv};
  const float*    bs[3] = {bq, bk, bv};
  float*          os[3] = {Qb, Kb, Vb};
  _Float16*       oh[3] = {Qh, Kh, Vh};
#pragma unroll
  for (int m = 0; m < 3; ++m) {
#pragma unroll
    for (int j = 0; j < 4; ++j) {
      v8f c = {};
      v16h b0 = load_b_rm(Ws[m], Dd, 0,  j * 16);
      v16h b1 = load_b_rm(Ws[m], Dd, 32, j * 16);
      c = wmma16(a0, b0, c);
      c = wmma16(a1, b1, c);
      const int col  = j * 16 + rloc;
      const float bi = bs[m][col];
#pragma unroll
      for (int r = 0; r < 8; ++r) {
        const int row = n0 + r + khalf * 8;
        const size_t idx = ((size_t)bt * Nn + row) * Dd + col;
        const float fv = c[r] + bi;
        os[m][idx] = fv;
        oh[m][idx] = (_Float16)fv;
      }
    }
  }
}

// ---------------------------------------------------------------------------
// Kernel 2: vp[bt,n] = V[bt,n,:] . Wp[:,0]
__global__ __launch_bounds__(256)
void vp_kernel(const float* __restrict__ V, const float* __restrict__ Wp,
               float* __restrict__ vp)
{
  const size_t i = (size_t)blockIdx.x * 256 + threadIdx.x; // over BT*N
  const float* vr = V + i * Dd;
  float acc = 0.f;
#pragma unroll
  for (int d = 0; d < Dd; ++d) acc += vr[d] * Wp[d];
  vp[i] = acc;
}

// ---------------------------------------------------------------------------
// Kernel 3: M[bt,n] = sum_k (Q[n].K[adj[n,k]]) * vp[adj[n,k]] + bp
// grid (N/256, BT), block 256.
__global__ __launch_bounds__(256)
void m_score(const float* __restrict__ Q, const float* __restrict__ K,
             const float* __restrict__ vp, const int* __restrict__ adj,
             const float* __restrict__ bp, float* __restrict__ M)
{
  const int bt = blockIdx.y;
  const int n  = blockIdx.x * 256 + threadIdx.x;
  float q[Dd];
  const float* qr = Q + ((size_t)bt * Nn + n) * Dd;
#pragma unroll
  for (int d = 0; d < Dd; ++d) q[d] = qr[d];
  float acc = bp[0];
  for (int k = 0; k < NNb; ++k) {
    const int j = adj[n * NNb + k];
    const float* kr = K + ((size_t)bt * Nn + j) * Dd;
    if (k + 1 < NNb) {
      const int jn = adj[n * NNb + k + 1];
      __builtin_prefetch(K + ((size_t)bt * Nn + jn) * Dd, 0, 1); // global_prefetch_b8
    }
    float d0 = 0.f;
#pragma unroll
    for (int d = 0; d < Dd; ++d) d0 += q[d] * kr[d];
    acc += d0 * vp[(size_t)bt * Nn + j];
  }
  M[(size_t)bt * Nn + n] = acc;
}

// ---------------------------------------------------------------------------
// Kernel 4: top-45 indices of M per (b,t) via 45 iterated LDS argmax rounds.
// grid BT, block 256.
__global__ __launch_bounds__(256)
void topk45(const float* __restrict__ M, int* __restrict__ mtop)
{
  __shared__ float mv[Nn];
  __shared__ float rv[256];
  __shared__ int   ri[256];
  const int bt  = blockIdx.x;
  const int tid = threadIdx.x;
  for (int i = tid; i < Nn; i += 256) mv[i] = M[(size_t)bt * Nn + i];
  __syncthreads();
  int last = 0;
  for (int it = 0; it < MS; ++it) {
    float bv = -3.4e38f; int bi = 0;
    for (int i = tid; i < Nn; i += 256) {
      const float v = mv[i];
      if (v > bv) { bv = v; bi = i; }
    }
    rv[tid] = bv; ri[tid] = bi;
    __syncthreads();
    for (int s = 128; s > 0; s >>= 1) {
      if (tid < s) {
        if (rv[tid + s] > rv[tid] ||
            (rv[tid + s] == rv[tid] && ri[tid + s] < ri[tid])) {
          rv[tid] = rv[tid + s]; ri[tid] = ri[tid + s];
        }
      }
      __syncthreads();
    }
    if (tid == 0) {
      last = ri[0];
      mtop[bt * MSP + it] = last;
      mv[last] = -3.4e38f;
    }
    __syncthreads();
  }
  if (tid == 0) {                 // pad rows 45..47 (excluded from argmax later)
    mtop[bt * MSP + 45] = last;
    mtop[bt * MSP + 46] = last;
    mtop[bt * MSP + 47] = last;
  }
}

// ---------------------------------------------------------------------------
// Kernel 5: scores = Q_red @ K^T / 8 ; softmax rows ; cp = per-col argmax(m<45);
//           av = attn @ V.  One block per (b,t), 8 waves.
// K/V slabs (f16) are async-staged into LDS; scores (48x512 f32) live in LDS.
// Dynamic LDS = 48*512*4 + 2*512*64*2 + 48*4  (~225 KB of the 320 KB WGP LDS).
// grid BT, block 256.
__global__ __launch_bounds__(256)
void attn_wmma(const _Float16* __restrict__ Qh, const _Float16* __restrict__ Kh,
               const _Float16* __restrict__ Vh, const int* __restrict__ mtop,
               int* __restrict__ cp, float* __restrict__ av)
{
  extern __shared__ float smem[];
  float*    sc = smem;                                   // [MSP][Nn] scores/attn
  _Float16* kh = (_Float16*)(smem + MSP * Nn);           // [Nn][Dd]
  _Float16* vh = kh + (size_t)Nn * Dd;                   // [Nn][Dd]
  int*      mt = (int*)(vh + (size_t)Nn * Dd);           // [MSP]

  const int bt    = blockIdx.x;
  const int wave  = threadIdx.x >> 5;
  const int lane  = threadIdx.x & 31;
  const int rloc  = lane & 15;
  const int khalf = lane >> 4;

  // Kick off async staging of this (b,t)'s K and V slabs (f16) into LDS.
  async_stage_slab(Kh + (size_t)bt * Nn * Dd, kh, threadIdx.x, 256);
  async_stage_slab(Vh + (size_t)bt * Nn * Dd, vh, threadIdx.x, 256);
  if (threadIdx.x < MSP) mt[threadIdx.x] = mtop[bt * MSP + threadIdx.x];
  wait_async_done();
  __syncthreads();

  // --- scores: 3 row-tiles x 32 col-tiles, K=64, B-frags straight from LDS ---
  for (int job = wave; job < 96; job += 8) {
    const int rt = job >> 5, ct = job & 31;
    const int gq = mt[rt * 16 + rloc];                 // gathered Q_red row
    const _Float16* qr = Qh + ((size_t)bt * Nn + gq) * Dd;
    const int kb = khalf * 8;
    v16h a0, a1;
#pragma unroll
    for (int e = 0; e < 8; ++e) {
      a0[e]     = qr[kb + e];
      a0[8 + e] = qr[kb + 16 + e];
      a1[e]     = qr[32 + kb + e];
      a1[8 + e] = qr[32 + kb + 16 + e];
    }
    // B = K^T: B[d][n] = K[n][d]  -> per-lane contiguous D-strip of LDS row n.
    const int ncol = ct * 16 + rloc;
    const _Float16* krow = kh + (size_t)ncol * Dd;
    const int kbB = khalf * 8;
    v16h b0, b1;
#pragma unroll
    for (int e = 0; e < 8; ++e) {
      b0[e]     = krow[kbB + e];
      b0[8 + e] = krow[kbB + 16 + e];
      b1[e]     = krow[32 + kbB + e];
      b1[8 + e] = krow[32 + kbB + 16 + e];
    }
    v8f c = {};
    c = wmma16(a0, b0, c);
    c = wmma16(a1, b1, c);
#pragma unroll
    for (int r = 0; r < 8; ++r)
      sc[(rt * 16 + r + khalf * 8) * Nn + ncol] = c[r] * 0.125f;
  }
  __syncthreads();

  // --- softmax: one wave per row ---
  for (int row = wave; row < MSP; row += 8) {
    float* sr = sc + (size_t)row * Nn;
    float mx = -3.4e38f;
    for (int j = lane; j < Nn; j += 32) mx = fmaxf(mx, sr[j]);
#pragma unroll
    for (int o = 16; o > 0; o >>= 1) mx = fmaxf(mx, __shfl_xor(mx, o, 32));
    float sum = 0.f;
    for (int j = lane; j < Nn; j += 32) {
      const float e = __expf(sr[j] - mx);
      sr[j] = e; sum += e;
    }
#pragma unroll
    for (int o = 16; o > 0; o >>= 1) sum += __shfl_xor(sum, o, 32);
    const float inv = 1.f / sum;
    for (int j = lane; j < Nn; j += 32) sr[j] *= inv;
  }
  __syncthreads();

  // --- cp[n] = argmax over m<45 of attn[m][n] (first max on ties) ---
  for (int n2 = threadIdx.x; n2 < Nn; n2 += 256) {
    float best = -3.4e38f; int bi = 0;
    for (int m = 0; m < MS; ++m) {
      const float v = sc[(size_t)m * Nn + n2];
      if (v > best) { best = v; bi = m; }
    }
    cp[(size_t)bt * Nn + n2] = bi;
  }

  // --- av = attn(48x512) @ V(512x64): 3 row-tiles x 4 d-tiles, 16 K-chunks ---
  for (int job = wave; job < 12; job += 8) {
    const int rt = job >> 2, dt = job & 3;
    const int rowm = rt * 16 + rloc;
    const int colD = dt * 16 + rloc;
    const int kbA = khalf * 8;
    const int kbB = khalf * 8;
    v8f c = {};
    for (int kc = 0; kc < 16; ++kc) {
      const int k0 = kc * 32;
      v16h a, b;
#pragma unroll
      for (int e = 0; e < 8; ++e) {
        a[e]     = (_Float16)sc[(size_t)rowm * Nn + k0 + kbA + e];
        a[8 + e] = (_Float16)sc[(size_t)rowm * Nn + k0 + kbA + 16 + e];
      }
#pragma unroll
      for (int e = 0; e < 8; ++e) {
        b[e]     = vh[(size_t)(k0 + kbB + e) * Dd + colD];
        b[8 + e] = vh[(size_t)(k0 + kbB + 16 + e) * Dd + colD];
      }
      c = wmma16(a, b, c);
    }
#pragma unroll
    for (int r = 0; r < 8; ++r)
      av[((size_t)bt * MSP + rt * 16 + r + khalf * 8) * Dd + colD] = c[r];
  }
}

// ---------------------------------------------------------------------------
// Kernel 6: value = LN(av[cp] @ Wo + bo + x_); out = LN(relu(value@W1+b1)@W2+b2 + value)
// grid (N/16, BT), block 32 (one wave per 16-row tile).
__global__ __launch_bounds__(32)
void out_wmma(const float* __restrict__ x, const float* __restrict__ eigvec,
              const float* __restrict__ eigval, const float* __restrict__ av,
              const int* __restrict__ cp,
              const float* __restrict__ Wo, const float* __restrict__ bo,
              const float* __restrict__ W1, const float* __restrict__ b1,
              const float* __restrict__ W2, const float* __restrict__ b2,
              float* __restrict__ out)
{
  __shared__ float sm [16 * LDSTR];   // value tile
  __shared__ float sm2[16 * LDSTR];   // h tile
  const int bt = blockIdx.y;
  const int n0 = blockIdx.x * 16;
  const int lane  = threadIdx.x;
  const int rloc  = lane & 15;
  const int khalf = lane >> 4;
  const int kb    = khalf * 8;

  // A fragments gathered from av rows cp[n]
  const int nrow = n0 + rloc;
  const int sel  = cp[(size_t)bt * Nn + nrow];
  const float* ar = av + ((size_t)bt * MSP + sel) * Dd;
  v16h a0, a1;
#pragma unroll
  for (int e = 0; e < 8; ++e) {
    a0[e]     = (_Float16)ar[kb + e];
    a0[8 + e] = (_Float16)ar[kb + 16 + e];
    a1[e]     = (_Float16)ar[32 + kb + e];
    a1[8 + e] = (_Float16)ar[32 + kb + 16 + e];
  }

  // value_pre = av[cp] @ Wo + bo + x_
#pragma unroll
  for (int j = 0; j < 4; ++j) {
    v8f c = {};
    c = wmma16(a0, load_b_rm(Wo, Dd, 0,  j * 16), c);
    c = wmma16(a1, load_b_rm(Wo, Dd, 32, j * 16), c);
    const int col = j * 16 + rloc;
#pragma unroll
    for (int r = 0; r < 8; ++r) {
      const int row = r + khalf * 8;
      const int ng  = n0 + row;
      const float xv = x[((size_t)bt * Nn + ng) * Dd + col]
                     + eigvec[(size_t)col * Nn + ng] * eigval[col];
      sm[row * LDSTR + col] = c[r] + bo[col] + xv;
    }
  }
  __syncthreads();
  // LayerNorm rows (lanes 0..15, one row each)
  if (lane < 16) {
    float mu = 0.f;
#pragma unroll
    for (int d = 0; d < Dd; ++d) mu += sm[lane * LDSTR + d];
    mu *= (1.f / Dd);
    float var = 0.f;
#pragma unroll
    for (int d = 0; d < Dd; ++d) { const float t = sm[lane * LDSTR + d] - mu; var += t * t; }
    const float inv = rsqrtf(var * (1.f / Dd) + 1e-5f);
#pragma unroll
    for (int d = 0; d < Dd; ++d) sm[lane * LDSTR + d] = (sm[lane * LDSTR + d] - mu) * inv;
  }
  __syncthreads();

  // h = relu(value @ W1 + b1)
  v16h va0, va1;
#pragma unroll
  for (int e = 0; e < 8; ++e) {
    va0[e]     = (_Float16)sm[rloc * LDSTR + kb + e];
    va0[8 + e] = (_Float16)sm[rloc * LDSTR + kb + 16 + e];
    va1[e]     = (_Float16)sm[rloc * LDSTR + 32 + kb + e];
    va1[8 + e] = (_Float16)sm[rloc * LDSTR + 32 + kb + 16 + e];
  }
#pragma unroll
  for (int j = 0; j < 4; ++j) {
    v8f c = {};
    c = wmma16(va0, load_b_rm(W1, Dd, 0,  j * 16), c);
    c = wmma16(va1, load_b_rm(W1, Dd, 32, j * 16), c);
    const int col = j * 16 + rloc;
#pragma unroll
    for (int r = 0; r < 8; ++r)
      sm2[(r + khalf * 8) * LDSTR + col] = fmaxf(c[r] + b1[col], 0.f);
  }
  __syncthreads();

  // h2 = h @ W2 + b2 + value  (A-frags read before sm2 is overwritten; LDS per-wave in-order)
  v16h ha0, ha1;
#pragma unroll
  for (int e = 0; e < 8; ++e) {
    ha0[e]     = (_Float16)sm2[rloc * LDSTR + kb + e];
    ha0[8 + e] = (_Float16)sm2[rloc * LDSTR + kb + 16 + e];
    ha1[e]     = (_Float16)sm2[rloc * LDSTR + 32 + kb + e];
    ha1[8 + e] = (_Float16)sm2[rloc * LDSTR + 32 + kb + 16 + e];
  }
  __syncthreads();
#pragma unroll
  for (int j = 0; j < 4; ++j) {
    v8f c = {};
    c = wmma16(ha0, load_b_rm(W2, Dd, 0,  j * 16), c);
    c = wmma16(ha1, load_b_rm(W2, Dd, 32, j * 16), c);
    const int col = j * 16 + rloc;
#pragma unroll
    for (int r = 0; r < 8; ++r) {
      const int row = r + khalf * 8;
      sm2[row * LDSTR + col] = c[r] + b2[col] + sm[row * LDSTR + col];
    }
  }
  __syncthreads();
  // final LN -> global out
  if (lane < 16) {
    float mu = 0.f;
#pragma unroll
    for (int d = 0; d < Dd; ++d) mu += sm2[lane * LDSTR + d];
    mu *= (1.f / Dd);
    float var = 0.f;
#pragma unroll
    for (int d = 0; d < Dd; ++d) { const float t = sm2[lane * LDSTR + d] - mu; var += t * t; }
    const float inv = rsqrtf(var * (1.f / Dd) + 1e-5f);
    float* orow = out + ((size_t)bt * Nn + n0 + lane) * Dd;
#pragma unroll
    for (int d = 0; d < Dd; ++d) orow[d] = (sm2[lane * LDSTR + d] - mu) * inv;
  }
}

// ---------------------------------------------------------------------------
extern "C" void kernel_launch(void* const* d_in, const int* in_sizes, int n_in,
                              void* d_out, int out_size, void* d_ws, size_t ws_size,
                              hipStream_t stream)
{
  const float* x      = (const float*)d_in[0];
  const int*   adj    = (const int*)  d_in[1];
  const float* eigvec = (const float*)d_in[2];
  const float* eigval = (const float*)d_in[3];
  const float* Wq = (const float*)d_in[4];  const float* bq = (const float*)d_in[5];
  const float* Wk = (const float*)d_in[6];  const float* bk = (const float*)d_in[7];
  const float* Wv = (const float*)d_in[8];  const float* bv = (const float*)d_in[9];
  const float* Wo = (const float*)d_in[10]; const float* bo = (const float*)d_in[11];
  const float* Wp = (const float*)d_in[12]; const float* bp = (const float*)d_in[13];
  const float* W1 = (const float*)d_in[14]; const float* b1 = (const float*)d_in[15];
  const float* W2 = (const float*)d_in[16]; const float* b2 = (const float*)d_in[17];
  float* out = (float*)d_out;

  // Workspace carve-up: f32 Q,K,V; vp; M; av; int mtop/cp; f16 Qh,Kh,Vh.
  float* ws  = (float*)d_ws;
  const size_t SZQ = (size_t)BT * Nn * Dd;
  float* Qb  = ws;
  float* Kb  = Qb + SZQ;
  float* Vb  = Kb + SZQ;
  float* vpb = Vb + SZQ;
  float* Mb  = vpb + (size_t)BT * Nn;
  float* avb = Mb  + (size_t)BT * Nn;
  int*   mtp = (int*)(avb + (size_t)BT * MSP * Dd);
  int*   cpb = mtp + (size_t)BT * MSP;
  _Float16* Qh = (_Float16*)(cpb + (size_t)BT * Nn);
  _Float16* Kh = Qh + SZQ;
  _Float16* Vh = Kh + SZQ;

  qkv_wmma<<<dim3(Nn / 16, BT), 32, 0, stream>>>(x, eigvec, eigval,
                                                 Wq, bq, Wk, bk, Wv, bv,
                                                 Qb, Kb, Vb, Qh, Kh, Vh);
  vp_kernel<<<(BT * Nn) / 256, 256, 0, stream>>>(Vb, Wp, vpb);
  m_score<<<dim3(Nn / 256, BT), 256, 0, stream>>>(Qb, Kb, vpb, adj, bp, Mb);
  topk45<<<BT, 256, 0, stream>>>(Mb, mtp);
  const size_t attn_lds = (size_t)MSP * Nn * sizeof(float)
                        + 2 * (size_t)Nn * Dd * sizeof(_Float16)
                        + MSP * sizeof(int);
  attn_wmma<<<BT, 256, attn_lds, stream>>>(Qh, Kh, Vh, mtp, cpb, avb);
  out_wmma<<<dim3(Nn / 16, BT), 32, 0, stream>>>(x, eigvec, eigval, avb, cpb,
                                                 Wo, bo, W1, b1, W2, b2, out);
}
